// PraxisAttention_57810259804538
// MI455X (gfx1250) — compile-verified, matrix operand order checked by reading, not run on hardware
//
#include <hip/hip_runtime.h>
#include <hip/hip_bf16.h>
#include <math.h>

typedef __bf16 bf16_t;
typedef __attribute__((ext_vector_type(16))) __bf16 v16bf;
typedef __attribute__((ext_vector_type(8)))  float  v8f;
typedef __attribute__((ext_vector_type(4)))  unsigned int v4u;
typedef __attribute__((ext_vector_type(8)))  int v8i;
typedef __attribute__((ext_vector_type(4)))  int v4i;

#define B_  2
#define S_  2048
#define DM_ 1024
#define H_  16
#define D_  64
#define LAMBDA_INIT 0.8f
#define GN_EPS 1e-5f

__device__ __forceinline__ v8f wmma_bf16(v16bf a, v16bf b, v8f c) {
  return __builtin_amdgcn_wmma_f32_16x16x32_bf16(false, a, false, b, (short)0, c,
                                                 false, false);
}

__device__ __forceinline__ v8f v8f_zero() {
  v8f z;
#pragma unroll
  for (int j = 0; j < 8; ++j) z[j] = 0.0f;
  return z;
}

// Fragment loader: 16 bf16 as two contiguous 8-element (16B) chunks at p and p+16.
__device__ __forceinline__ v16bf load_frag(const bf16_t* p) {
  union { uint4 u[2]; v16bf v; } t;
  t.u[0] = *(const uint4*)p;
  t.u[1] = *(const uint4*)(p + 16);
  return t.v;
}

// LDS byte offset of a generic shared pointer: flat LDS-aperture addresses carry
// the wave-relative LDS offset in addr[31:0] (ISA 10.2).
__device__ __forceinline__ unsigned lds_off(const void* p) {
  return (unsigned)(unsigned long long)p;
}

// Issue a TDM 2D tile load: tile_w x tile_h elements (bf16), global row stride
// row_stride elements, packed contiguously into LDS at lptr. One DMA per call
// (per-wave instruction; EXEC ignored). Tracked by TENSORcnt.
__device__ __forceinline__ void tdm_load_2d(const void* gptr, void* lptr,
                                            unsigned tile_w, unsigned tile_h,
                                            unsigned row_stride) {
  unsigned long long ga = (unsigned long long)gptr;
  v4u g0;
  g0[0] = 1u;                                   // count=1, user mode, no gather
  g0[1] = lds_off(lptr);                        // lds_addr
  g0[2] = (unsigned)(ga & 0xFFFFFFFFu);         // global_addr[31:0]
  g0[3] = (unsigned)((ga >> 32) & 0x1FFFFFFu)   // global_addr[56:32]
          | 0x80000000u;                        // type=2 ("image")
  unsigned ge[8];
  ge[0] = 1u << 16;                             // data_size=2B; no multicast/pad
  ge[1] = (tile_w & 0xFFFFu) << 16;             // tensor_dim0 lo16 (== tile_w)
  ge[2] = (tile_w >> 16) | ((tile_h & 0xFFFFu) << 16);  // td0 hi | tensor_dim1 lo
  ge[3] = (tile_h >> 16) | ((tile_w & 0xFFFFu) << 16);  // td1 hi | tile_dim0
  ge[4] = tile_h & 0xFFFFu;                     // tile_dim1 (tile_dim2 = 0)
  ge[5] = row_stride;                           // tensor_dim0_stride lo32
  ge[6] = 0;                                    // stride hi16 | dim1_stride lo16
  ge[7] = 0;
  v8i g1;
#pragma unroll
  for (int j = 0; j < 8; ++j) g1[j] = (int)ge[j];
  v4i z4;
#pragma unroll
  for (int j = 0; j < 4; ++j) z4[j] = 0;
  v8i z8;
#pragma unroll
  for (int j = 0; j < 8; ++j) z8[j] = 0;
  __builtin_amdgcn_tensor_load_to_lds(g0, g1, z4, z4, z8, 0);
}

// ---------------------------------------------------------------- converts
__global__ void cvt_f32_bf16(const float* __restrict__ in, bf16_t* __restrict__ out, int n) {
  int i = blockIdx.x * blockDim.x + threadIdx.x;
  if (i < n) out[i] = (bf16_t)in[i];
}

// ---------------------------------------------------------------- lambda
__global__ void lambda_kernel(const float* __restrict__ lq1, const float* __restrict__ lq2,
                              const float* __restrict__ lk1, const float* __restrict__ lk2,
                              float* __restrict__ lam) {
  int l = threadIdx.x;  // 32 threads, one wave
  float d1 = lq1[l] * lk1[l] + lq1[l + 32] * lk1[l + 32];
  float d2 = lq2[l] * lk2[l] + lq2[l + 32] * lk2[l + 32];
#pragma unroll
  for (int off = 16; off; off >>= 1) {
    d1 += __shfl_xor(d1, off, 32);
    d2 += __shfl_xor(d2, off, 32);
  }
  if (l == 0) *lam = __expf(d1) - __expf(d2) + LAMBDA_INIT;
}

// ---------------------------------------------------------------- GEMM (bf16 x bf16 -> OutT)
// C[M,N] = A[M,K] * W[K,N], all row-major. Block tile 128x64, K-step 32.
// A tile staged by TDM (tensor_load_to_lds); W tile staged transposed by VALU path.
template <typename OutT>
__global__ __launch_bounds__(256) void wmma_gemm(const bf16_t* __restrict__ A,
                                                 const bf16_t* __restrict__ W,
                                                 OutT* __restrict__ C,
                                                 int M, int N, int K) {
  __shared__ bf16_t As[128][32];   // 8 KB  (TDM destination, row-major fill)
  __shared__ bf16_t Bst[64][32];   // 4 KB  (transposed: [col][k])

  const int tid  = threadIdx.x;
  const int wv   = tid >> 5;
  const int lane = tid & 31;
  const int wm   = wv & 3;        // 0..3
  const int wn   = wv >> 2;       // 0..1
  const int bm   = blockIdx.x * 128;
  const int bn   = blockIdx.y * 64;
  const int c    = lane & 15;
  const int kb   = (lane >> 4) * 8;  // 0 or 8

  v8f acc[2][2];
#pragma unroll
  for (int mt = 0; mt < 2; ++mt)
#pragma unroll
    for (int nt = 0; nt < 2; ++nt) acc[mt][nt] = v8f_zero();

  const int brow = tid >> 3, bseg = (tid & 7) * 8;

  for (int k0 = 0; k0 < K; k0 += 32) {
    __syncthreads();
    if (wv == 0)  // DMA the 128x32 A tile straight into LDS
      tdm_load_2d(A + (size_t)bm * K + k0, &As[0][0], 32u, 128u, (unsigned)K);
    {  // stage W tile (32x64) transposed into Bst[col][k]
      union { uint4 u; bf16_t e[8]; } t;
      const bf16_t* src = W + (size_t)(k0 + brow) * N + bn + bseg;
      t.u = *(const uint4*)src;
#pragma unroll
      for (int j = 0; j < 8; ++j) Bst[bseg + j][brow] = t.e[j];
      if (k0 + 32 < K) __builtin_prefetch(src + (size_t)32 * N, 0, 1);
    }
    if (wv == 0) __builtin_amdgcn_s_wait_tensorcnt(0);
    __syncthreads();

    v16bf af[2], bfv[2];
#pragma unroll
    for (int mt = 0; mt < 2; ++mt)
      af[mt] = load_frag(&As[wm * 32 + mt * 16 + c][kb]);
#pragma unroll
    for (int nt = 0; nt < 2; ++nt)
      bfv[nt] = load_frag(&Bst[wn * 32 + nt * 16 + c][kb]);
#pragma unroll
    for (int mt = 0; mt < 2; ++mt)
#pragma unroll
      for (int nt = 0; nt < 2; ++nt) acc[mt][nt] = wmma_bf16(af[mt], bfv[nt], acc[mt][nt]);
  }

  const int hi8 = kb;  // C-layout row offset per lane half
#pragma unroll
  for (int mt = 0; mt < 2; ++mt)
#pragma unroll
    for (int nt = 0; nt < 2; ++nt)
#pragma unroll
      for (int i = 0; i < 8; ++i) {
        int row = bm + wm * 32 + mt * 16 + hi8 + i;
        int col = bn + wn * 32 + nt * 16 + c;
        C[(size_t)row * N + col] = (OutT)acc[mt][nt][i];
      }
}

// ---------------------------------------------------------------- differential flash attention
// grid = B*H*(S/64) blocks of 128 threads (4 waves, 16 q-rows per wave).
// K tile (32x128, both halves) staged by TDM; V staged transposed Vt[d][key].
__global__ __launch_bounds__(128) void diff_attn_kernel(
    const bf16_t* __restrict__ qg, const bf16_t* __restrict__ kg,
    const bf16_t* __restrict__ vg, const int* __restrict__ tok,
    const float* __restrict__ amask, const float* __restrict__ lam_p,
    float* __restrict__ og) {
  const int qb = blockIdx.x & 31;
  const int h  = (blockIdx.x >> 5) & 15;
  const int b  = blockIdx.x >> 9;
  const int tid  = threadIdx.x;
  const int wv   = tid >> 5;
  const int lane = tid & 31;
  const int c    = lane & 15;
  const int kb   = (lane >> 4) * 8;  // 0 or 8
  const int hi8  = kb;

  __shared__ bf16_t Ks[32][128];        // [key][d: 0..63 = K1, 64..127 = K2]  8 KB
  __shared__ bf16_t Vt[64][32];         // [d][key]                            4 KB
  __shared__ bf16_t Pb[4][2][16][32];   // per-wave prob scratch               8 KB
  __shared__ float  kposf[32];
  __shared__ float  kmaskf[32];

  const float slope = exp2f(-8.0f * (float)(h + 1) / 16.0f);
  const float rec   = 0.125f;  // 1/sqrt(64)
  const float lam   = *lam_p;

  // Q fragments (A-layout, 16x32 per 32-d chunk), both halves
  v16bf aq1[2], aq2[2];
  {
    const size_t rowoff = (size_t)(b * S_ + qb * 64 + wv * 16 + c) * (size_t)(2 * H_ * D_);
#pragma unroll
    for (int dc = 0; dc < 2; ++dc) {
      aq1[dc] = load_frag(qg + rowoff + h * 128 + dc * 32 + kb);
      aq2[dc] = load_frag(qg + rowoff + h * 128 + 64 + dc * 32 + kb);
    }
  }

  const int qbase = qb * 64 + wv * 16 + hi8;
  float qposf[8];
#pragma unroll
  for (int i = 0; i < 8; ++i) qposf[i] = (float)tok[b * S_ + qbase + i];

  float m1[8], l1[8], m2[8], l2[8];
  v8f o1[4], o2[4];
#pragma unroll
  for (int i = 0; i < 8; ++i) { m1[i] = -1e30f; m2[i] = -1e30f; l1[i] = 0.f; l2[i] = 0.f; }
#pragma unroll
  for (int nt = 0; nt < 4; ++nt) { o1[nt] = v8f_zero(); o2[nt] = v8f_zero(); }

  const int kt_last = 2 * qb + 1;  // causal: keys <= qb*64+63
  for (int kt = 0; kt <= kt_last; ++kt) {
    __syncthreads();
    const size_t krow0 = (size_t)(b * S_ + kt * 32);
    if (wv == 0)  // DMA the 32x128 K tile (both halves) into Ks
      tdm_load_2d(kg + krow0 * (2 * H_ * D_) + h * 128, &Ks[0][0],
                  128u, 32u, (unsigned)(2 * H_ * D_));
    {  // V: transpose-scatter 16 contiguous d into Vt columns (overlaps DMA)
      const int kidx = tid >> 2;
      const int dseg = (tid & 3) << 4;  // 0,16,32,48
      union { uint4 u[2]; bf16_t e[16]; } t;
      const bf16_t* pv = vg + (krow0 + kidx) * (H_ * D_) + h * 64 + dseg;
      t.u[0] = *(const uint4*)pv;
      t.u[1] = *(const uint4*)(pv + 8);
#pragma unroll
      for (int j = 0; j < 16; ++j) Vt[dseg + j][kidx] = t.e[j];
    }
    if (tid < 32) {
      kposf[tid]  = (float)tok[b * S_ + kt * 32 + tid];
      kmaskf[tid] = (1.0f - amask[b * S_ + kt * 32 + tid]) * -1e9f;
    }
    if (wv == 0) __builtin_amdgcn_s_wait_tensorcnt(0);
    __syncthreads();

    // V B-fragments (shared across both softmax branches): contiguous b128 loads
    v16bf vb[4];
#pragma unroll
    for (int nt = 0; nt < 4; ++nt) vb[nt] = load_frag(&Vt[nt * 16 + c][kb]);

    // scores: two 16x16 tiles per branch (keys 0-15, 16-31)
    v8f s10 = v8f_zero(), s11 = v8f_zero(), s20 = v8f_zero(), s21 = v8f_zero();
#pragma unroll
    for (int dc = 0; dc < 2; ++dc) {
      v16bf bk0 = load_frag(&Ks[c][dc * 32 + kb]);
      v16bf bk1 = load_frag(&Ks[16 + c][dc * 32 + kb]);
      s10 = wmma_bf16(aq1[dc], bk0, s10);
      s11 = wmma_bf16(aq1[dc], bk1, s11);
      bk0 = load_frag(&Ks[c][64 + dc * 32 + kb]);
      bk1 = load_frag(&Ks[16 + c][64 + dc * 32 + kb]);
      s20 = wmma_bf16(aq2[dc], bk0, s20);
      s21 = wmma_bf16(aq2[dc], bk1, s21);
    }

    // online softmax for both branches; P written to LDS for C->A relayout
    const int key0 = kt * 32;
    float r1[8], r2[8];
#pragma unroll
    for (int i = 0; i < 8; ++i) {
      const int qi = qbase + i;
      const float add0 = -slope * (qposf[i] - kposf[c]) +
                         ((key0 + c > qi) ? -1e9f : 0.f) + kmaskf[c];
      const float add1 = -slope * (qposf[i] - kposf[16 + c]) +
                         ((key0 + 16 + c > qi) ? -1e9f : 0.f) + kmaskf[16 + c];
      {  // branch 1
        float x0 = s10[i] * rec + add0, x1 = s11[i] * rec + add1;
        float rmax = fmaxf(x0, x1);
#pragma unroll
        for (int off = 8; off; off >>= 1) rmax = fmaxf(rmax, __shfl_xor(rmax, off, 16));
        float mn = fmaxf(m1[i], rmax);
        float corr = __expf(m1[i] - mn);
        float p0 = __expf(x0 - mn), p1v = __expf(x1 - mn);
        float rs = p0 + p1v;
#pragma unroll
        for (int off = 8; off; off >>= 1) rs += __shfl_xor(rs, off, 16);
        l1[i] = l1[i] * corr + rs; m1[i] = mn; r1[i] = corr;
        Pb[wv][0][hi8 + i][c] = (bf16_t)p0;
        Pb[wv][0][hi8 + i][16 + c] = (bf16_t)p1v;
      }
      {  // branch 2
        float x0 = s20[i] * rec + add0, x1 = s21[i] * rec + add1;
        float rmax = fmaxf(x0, x1);
#pragma unroll
        for (int off = 8; off; off >>= 1) rmax = fmaxf(rmax, __shfl_xor(rmax, off, 16));
        float mn = fmaxf(m2[i], rmax);
        float corr = __expf(m2[i] - mn);
        float p0 = __expf(x0 - mn), p1v = __expf(x1 - mn);
        float rs = p0 + p1v;
#pragma unroll
        for (int off = 8; off; off >>= 1) rs += __shfl_xor(rs, off, 16);
        l2[i] = l2[i] * corr + rs; m2[i] = mn; r2[i] = corr;
        Pb[wv][1][hi8 + i][c] = (bf16_t)p0;
        Pb[wv][1][hi8 + i][16 + c] = (bf16_t)p1v;
      }
    }
#pragma unroll
    for (int nt = 0; nt < 4; ++nt)
#pragma unroll
      for (int i = 0; i < 8; ++i) { o1[nt][i] *= r1[i]; o2[nt][i] *= r2[i]; }

    __syncthreads();

    {  // O += P * V for both branches (A-frags are contiguous b128 loads)
      v16bf ap = load_frag(&Pb[wv][0][c][kb]);
#pragma unroll
      for (int nt = 0; nt < 4; ++nt) o1[nt] = wmma_bf16(ap, vb[nt], o1[nt]);
      ap = load_frag(&Pb[wv][1][c][kb]);
#pragma unroll
      for (int nt = 0; nt < 4; ++nt) o2[nt] = wmma_bf16(ap, vb[nt], o2[nt]);
    }
  }

  // epilogue: normalize, combine branches, write o [B,H,S,D] fp32
#pragma unroll
  for (int nt = 0; nt < 4; ++nt)
#pragma unroll
    for (int i = 0; i < 8; ++i) {
      float w1 = o1[nt][i] / l1[i];
      float w2 = o2[nt][i] / l2[i];
      int row = qbase + i;
      int d   = nt * 16 + c;
      og[((size_t)(b * H_ + h) * S_ + row) * D_ + d] = w1 - lam * w2;
    }
}

// ---------------------------------------------------------------- GroupNorm stats (per b,h)
__global__ __launch_bounds__(256) void gn_stats_kernel(const float* __restrict__ o,
                                                       float* __restrict__ stats) {
  const int bh = blockIdx.x;
  const float* p = o + (size_t)bh * S_ * D_;
  float s = 0.f, s2 = 0.f;
  for (int i = threadIdx.x; i < S_ * D_; i += 256) {
    float v = p[i]; s += v; s2 += v * v;
  }
  __shared__ float sh0[256], sh1[256];
  sh0[threadIdx.x] = s; sh1[threadIdx.x] = s2;
  __syncthreads();
  for (int st = 128; st; st >>= 1) {
    if (threadIdx.x < st) { sh0[threadIdx.x] += sh0[threadIdx.x + st]; sh1[threadIdx.x] += sh1[threadIdx.x + st]; }
    __syncthreads();
  }
  if (threadIdx.x == 0) {
    float inv = 1.0f / (float)(S_ * D_);
    float mean = sh0[0] * inv;
    float var  = sh1[0] * inv - mean * mean;
    stats[bh * 2]     = mean;
    stats[bh * 2 + 1] = rsqrtf(var + GN_EPS);
  }
}

// ---------------------------------------------------------------- GroupNorm apply -> bf16 [B,S,H*D]
__global__ void gn_apply_kernel(const float* __restrict__ o, const float* __restrict__ stats,
                                const float* __restrict__ gw, const float* __restrict__ gb,
                                bf16_t* __restrict__ out) {
  int idx = blockIdx.x * blockDim.x + threadIdx.x;
  if (idx >= B_ * H_ * S_ * D_) return;
  int d   = idx & 63;
  int row = (idx >> 6) & (S_ - 1);
  int bh  = idx >> 17;
  int h   = bh & 15;
  int b   = bh >> 4;
  float mean = stats[bh * 2], rstd = stats[bh * 2 + 1];
  float v = (o[idx] - mean) * rstd * gw[h * 64 + d] + gb[h * 64 + d];
  v *= (1.0f - LAMBDA_INIT);
  out[(size_t)(b * S_ + row) * DM_ + h * 64 + d] = (bf16_t)v;
}

// ---------------------------------------------------------------- launch
extern "C" void kernel_launch(void* const* d_in, const int* in_sizes, int n_in,
                              void* d_out, int out_size, void* d_ws, size_t ws_size,
                              hipStream_t stream) {
  (void)in_sizes; (void)n_in; (void)out_size; (void)ws_size;
  const float* inputs = (const float*)d_in[0];
  const float* amask  = (const float*)d_in[1];
  const int*   tok    = (const int*)d_in[2];
  const float* Wq  = (const float*)d_in[3];
  const float* Wk  = (const float*)d_in[4];
  const float* Wv  = (const float*)d_in[5];
  const float* Wo  = (const float*)d_in[6];
  const float* lq1 = (const float*)d_in[7];
  const float* lq2 = (const float*)d_in[8];
  const float* lk1 = (const float*)d_in[9];
  const float* lk2 = (const float*)d_in[10];
  const float* gnw = (const float*)d_in[11];
  const float* gnb = (const float*)d_in[12];
  float* out = (float*)d_out;

  char* ws = (char*)d_ws;
  size_t off = 0;
  auto alloc = [&](size_t bytes) -> char* {
    char* p = ws + off;
    off += (bytes + 255) & ~(size_t)255;
    return p;
  };
  const size_t MROWS = (size_t)B_ * S_;  // 4096
  bf16_t* in_bf  = (bf16_t*)alloc(MROWS * DM_ * 2);
  bf16_t* Wq_bf  = (bf16_t*)alloc((size_t)DM_ * 2048 * 2);
  bf16_t* Wk_bf  = (bf16_t*)alloc((size_t)DM_ * 2048 * 2);
  bf16_t* Wv_bf  = (bf16_t*)alloc((size_t)DM_ * 1024 * 2);
  bf16_t* Wo_bf  = (bf16_t*)alloc((size_t)DM_ * 1024 * 2);
  bf16_t* q_bf   = (bf16_t*)alloc(MROWS * 2048 * 2);
  bf16_t* k_bf   = (bf16_t*)alloc(MROWS * 2048 * 2);
  bf16_t* v_bf   = (bf16_t*)alloc(MROWS * 1024 * 2);
  float*  o_ws   = (float*)alloc((size_t)B_ * H_ * S_ * D_ * 4);
  bf16_t* on_bf  = (bf16_t*)alloc(MROWS * DM_ * 2);
  float*  stats  = (float*)alloc((size_t)B_ * H_ * 2 * 4);
  float*  lam_ws = (float*)alloc(4);

  int n;
  n = (int)(MROWS * DM_);
  cvt_f32_bf16<<<(n + 255) / 256, 256, 0, stream>>>(inputs, in_bf, n);
  n = DM_ * 2048;
  cvt_f32_bf16<<<(n + 255) / 256, 256, 0, stream>>>(Wq, Wq_bf, n);
  cvt_f32_bf16<<<(n + 255) / 256, 256, 0, stream>>>(Wk, Wk_bf, n);
  n = DM_ * 1024;
  cvt_f32_bf16<<<(n + 255) / 256, 256, 0, stream>>>(Wv, Wv_bf, n);
  cvt_f32_bf16<<<(n + 255) / 256, 256, 0, stream>>>(Wo, Wo_bf, n);
  lambda_kernel<<<1, 32, 0, stream>>>(lq1, lq2, lk1, lk2, lam_ws);

  dim3 blk(256);
  wmma_gemm<bf16_t><<<dim3(32, 32), blk, 0, stream>>>(in_bf, Wq_bf, q_bf, 4096, 2048, 1024);
  wmma_gemm<bf16_t><<<dim3(32, 32), blk, 0, stream>>>(in_bf, Wk_bf, k_bf, 4096, 2048, 1024);
  wmma_gemm<bf16_t><<<dim3(32, 16), blk, 0, stream>>>(in_bf, Wv_bf, v_bf, 4096, 1024, 1024);

  diff_attn_kernel<<<B_ * H_ * (S_ / 64), 128, 0, stream>>>(q_bf, k_bf, v_bf, tok, amask,
                                                            lam_ws, o_ws);

  gn_stats_kernel<<<B_ * H_, 256, 0, stream>>>(o_ws, stats);
  n = B_ * H_ * S_ * D_;
  gn_apply_kernel<<<(n + 255) / 256, 256, 0, stream>>>(o_ws, stats, gnw, gnb, on_bf);

  wmma_gemm<float><<<dim3(32, 16), blk, 0, stream>>>(on_bf, Wo_bf, out, 4096, 1024, 1024);
}
